// gaeGAT_layer_80685255622922
// MI455X (gfx1250) — compile-verified, matrix operand order checked by reading, not run on hardware
//
#include <hip/hip_runtime.h>

// ---------------------------------------------------------------------------
// GAT layer, fused for MI455X (gfx1250, wave32).
//   H   = X @ W^T + b                      (f32 WMMA 16x16x4)
//   s,r = H @ a_s, H @ a_r
//   p_ij = A_ij * exp(lrelu(s_i + r_j) - mhat_i),  mhat_i = lrelu(s_i + Rmax)
//   out_i = (sum_j p_ij * H_j) / (sum_j p_ij)      (f16 WMMA 16x16x32, f32 acc)
// Softmax shift-invariance + per-row upper bound mhat_i removes the row-max
// pass, so the 268 MB adjacency streams through exactly ONCE (NT loads).
// ---------------------------------------------------------------------------

typedef float    f32x2  __attribute__((ext_vector_type(2)));
typedef float    f32x4  __attribute__((ext_vector_type(4)));
typedef float    f32x8  __attribute__((ext_vector_type(8)));
typedef _Float16 f16x8  __attribute__((ext_vector_type(8)));
typedef _Float16 f16x16 __attribute__((ext_vector_type(16)));

#define NND   8192
#define DIN   128
#define DOUT  128

// ---------------------------------------------------------------------------
// Kernel 1: H = X @ W^T + b  via v_wmma_f32_16x16x4_f32.
// Block = 256 threads = 8 waves; each wave owns a 16-row strip, loops 8 col
// chunks of 16. A-frag (16x4 f32): lane<16 -> M=lane, K={4k,4k+1};
// lane>=16 -> K={4k+2,4k+3}. B-frag mirrors with N instead of M; B[k][n]=W[n][k].
// Also emits HhT = H^T in f16 so k_attn's B fragments are contiguous b128s.
// ---------------------------------------------------------------------------
__global__ __launch_bounds__(256) void k_linear(
    const float* __restrict__ X, const float* __restrict__ W,
    const float* __restrict__ b, float* __restrict__ H,
    _Float16* __restrict__ HhT)
{
    const int lane  = threadIdx.x & 31;
    const int wave  = threadIdx.x >> 5;
    const int lrow  = lane & 15;
    const int khalf = lane >> 4;
    const int i0    = blockIdx.x * 128 + wave * 16;

    // preload all 32 K-fragments of the X strip (64 VGPRs)
    f32x2 af[32];
    const float* xrow = X + (size_t)(i0 + lrow) * DIN + khalf * 2;
#pragma unroll
    for (int kk = 0; kk < 32; ++kk)
        af[kk] = *(const f32x2*)(xrow + kk * 4);

    for (int c = 0; c < 8; ++c) {
        f32x8 acc = (f32x8){0.f,0.f,0.f,0.f,0.f,0.f,0.f,0.f};
        const int n = c * 16 + lrow;
        const float* wrow = W + (size_t)n * DIN + khalf * 2;
#pragma unroll
        for (int kk = 0; kk < 32; ++kk) {
            f32x2 bf = *(const f32x2*)(wrow + kk * 4);
            acc = __builtin_amdgcn_wmma_f32_16x16x4_f32(
                false, af[kk], false, bf, (short)0, acc, false, false);
        }
        const float bias = b[n];
#pragma unroll
        for (int rr = 0; rr < 8; ++rr) {
            const int row = i0 + rr + 8 * khalf;   // D layout: M = rr + 8*(lane>=16)
            const float v = acc[rr] + bias;
            H[(size_t)row * DOUT + n]   = v;
            HhT[(size_t)n * NND + row]  = (_Float16)v;
        }
    }
}

// ---------------------------------------------------------------------------
// Kernel 2: s_i = H_i . a_s ; r_i = H_i . a_r  (one 128-thread block per row)
// ---------------------------------------------------------------------------
__global__ __launch_bounds__(128) void k_scores(
    const float* __restrict__ H, const float* __restrict__ a_s,
    const float* __restrict__ a_r, float* __restrict__ s, float* __restrict__ r)
{
    __shared__ float red[8];
    const int i = blockIdx.x, t = threadIdx.x;
    const float h  = H[(size_t)i * DOUT + t];
    float ps = h * a_s[t];
    float pr = h * a_r[t];
#pragma unroll
    for (int off = 16; off > 0; off >>= 1) {
        ps += __shfl_down(ps, off, 32);
        pr += __shfl_down(pr, off, 32);
    }
    const int w = t >> 5;
    if ((t & 31) == 0) { red[w] = ps; red[4 + w] = pr; }
    __syncthreads();
    if (t == 0) s[i] = red[0] + red[1] + red[2] + red[3];
    if (t == 1) r[i] = red[4] + red[5] + red[6] + red[7];
}

// ---------------------------------------------------------------------------
// Kernel 3: Rmax = max_j r_j  (single block)
// ---------------------------------------------------------------------------
__global__ __launch_bounds__(256) void k_rmax(
    const float* __restrict__ r, float* __restrict__ rmax)
{
    __shared__ float red[256];
    float m = -3.4e38f;
    for (int j = threadIdx.x; j < NND; j += 256) m = fmaxf(m, r[j]);
    red[threadIdx.x] = m;
    __syncthreads();
    for (int off = 128; off > 0; off >>= 1) {
        if (threadIdx.x < off)
            red[threadIdx.x] = fmaxf(red[threadIdx.x], red[threadIdx.x + off]);
        __syncthreads();
    }
    if (threadIdx.x == 0) rmax[0] = red[0];
}

// ---------------------------------------------------------------------------
// Kernel 4: fused mask/softmax/att@H. Block = 128 threads = 4 waves owning one
// 16-row strip; waves stripe 256 j-tiles of K=32. P built in-register in the
// 16-bit WMMA A-fragment layout; B frags are b128s from HhT. A streamed NT.
// ---------------------------------------------------------------------------
__global__ __launch_bounds__(128) void k_attn(
    const float* __restrict__ A, const float* __restrict__ s,
    const float* __restrict__ r, const float* __restrict__ rmax_p,
    const _Float16* __restrict__ HhT, float* __restrict__ out)
{
    __shared__ float accbuf[16 * DOUT];
    __shared__ float lsumbuf[16];
    const int lane  = threadIdx.x & 31;
    const int wave  = threadIdx.x >> 5;
    const int lrow  = lane & 15;
    const int khalf = lane >> 4;
    const int i0    = blockIdx.x * 16;

    for (int t = threadIdx.x; t < 16 * DOUT; t += 128) accbuf[t] = 0.f;
    if (threadIdx.x < 16) lsumbuf[threadIdx.x] = 0.f;
    __syncthreads();

    const float Rmax = rmax_p[0];
    const float sv   = s[i0 + lrow];
    const float tb   = sv + Rmax;
    const float mhat = tb > 0.f ? tb : 0.2f * tb;   // per-row softmax shift (>= row max)
    const float L2E  = 1.44269504088896340736f;

    f32x8 acc[8];
#pragma unroll
    for (int c = 0; c < 8; ++c) acc[c] = (f32x8){0.f,0.f,0.f,0.f,0.f,0.f,0.f,0.f};
    float lsum = 0.f;

    const float* arow = A + (size_t)(i0 + lrow) * NND;

    for (int jt = wave; jt < NND / 32; jt += 4) {
        const int j0  = jt * 32;
        const int kb1 = j0 + khalf * 8;        // this lane's K 0..7  (or 8..15)
        const int kb2 = j0 + 16 + khalf * 8;   // this lane's K 16..23 (or 24..31)

        const f32x4 a0 = __builtin_nontemporal_load((const f32x4*)(arow + kb1));
        const f32x4 a1 = __builtin_nontemporal_load((const f32x4*)(arow + kb1 + 4));
        const f32x4 a2 = __builtin_nontemporal_load((const f32x4*)(arow + kb2));
        const f32x4 a3 = __builtin_nontemporal_load((const f32x4*)(arow + kb2 + 4));
        const f32x4 r0 = *(const f32x4*)(r + kb1);
        const f32x4 r1 = *(const f32x4*)(r + kb1 + 4);
        const f32x4 r2 = *(const f32x4*)(r + kb2);
        const f32x4 r3 = *(const f32x4*)(r + kb2 + 4);

        f16x16 pa;
#pragma unroll
        for (int t = 0; t < 4; ++t) {
            float e, p;
            e = sv + r0[t]; e = e > 0.f ? e : 0.2f * e;
            p = a0[t] * __builtin_amdgcn_exp2f((e - mhat) * L2E);
            lsum += p; pa[t] = (_Float16)p;
            e = sv + r1[t]; e = e > 0.f ? e : 0.2f * e;
            p = a1[t] * __builtin_amdgcn_exp2f((e - mhat) * L2E);
            lsum += p; pa[4 + t] = (_Float16)p;
            e = sv + r2[t]; e = e > 0.f ? e : 0.2f * e;
            p = a2[t] * __builtin_amdgcn_exp2f((e - mhat) * L2E);
            lsum += p; pa[8 + t] = (_Float16)p;
            e = sv + r3[t]; e = e > 0.f ? e : 0.2f * e;
            p = a3[t] * __builtin_amdgcn_exp2f((e - mhat) * L2E);
            lsum += p; pa[12 + t] = (_Float16)p;
        }

#pragma unroll
        for (int c = 0; c < 8; ++c) {
            const _Float16* hrow = HhT + (size_t)(c * 16 + lrow) * NND;
            const f16x8 blo = *(const f16x8*)(hrow + kb1);
            const f16x8 bhi = *(const f16x8*)(hrow + kb2);
            const f16x16 bf = __builtin_shufflevector(
                blo, bhi, 0,1,2,3,4,5,6,7,8,9,10,11,12,13,14,15);
            acc[c] = __builtin_amdgcn_wmma_f32_16x16x32_f16(
                false, pa, false, bf, (short)0, acc[c], false, false);
        }
    }

    // combine lane halves (disjoint K coverage, same row), then cross-wave in LDS
    lsum += __shfl_xor(lsum, 16, 32);
#pragma unroll
    for (int c = 0; c < 8; ++c)
#pragma unroll
        for (int rr = 0; rr < 8; ++rr) {
            const int row = rr + 8 * khalf;
            const int col = c * 16 + lrow;
            atomicAdd(&accbuf[row * DOUT + col], acc[c][rr]);
        }
    if (lane < 16) atomicAdd(&lsumbuf[lrow], lsum);
    __syncthreads();

    for (int t = threadIdx.x; t < 16 * DOUT; t += 128) {
        const int m = t >> 7, n = t & 127;
        out[(size_t)(i0 + m) * DOUT + n] = accbuf[t] / lsumbuf[m];
    }
}

// ---------------------------------------------------------------------------
// Launcher. Inputs: X[8192*128], A[8192*8192], W[128*128], b[128],
// a_s[128], a_r[128], all f32. Output: H_out[8192*128] f32.
// Workspace layout: H f32 (4MB) | HhT f16 (2MB) | s (32KB) | r (32KB) | Rmax
// ---------------------------------------------------------------------------
extern "C" void kernel_launch(void* const* d_in, const int* in_sizes, int n_in,
                              void* d_out, int out_size, void* d_ws, size_t ws_size,
                              hipStream_t stream) {
    const float* X   = (const float*)d_in[0];
    const float* A   = (const float*)d_in[1];
    const float* W   = (const float*)d_in[2];
    const float* b   = (const float*)d_in[3];
    const float* a_s = (const float*)d_in[4];
    const float* a_r = (const float*)d_in[5];
    float* out = (float*)d_out;

    char* ws = (char*)d_ws;
    float*    H    = (float*)ws;                                   // 4 MB
    _Float16* HhT  = (_Float16*)(ws + (size_t)4 * 1024 * 1024);    // 2 MB
    float*    sbuf = (float*)(ws + (size_t)6 * 1024 * 1024);       // 32 KB
    float*    rbuf = sbuf + NND;                                   // 32 KB
    float*    rmax = rbuf + NND;                                   // 4 B

    k_linear<<<NND / 128, 256, 0, stream>>>(X, W, b, H, HhT);
    k_scores<<<NND, 128, 0, stream>>>(H, a_s, a_r, sbuf, rbuf);
    k_rmax  <<<1, 256, 0, stream>>>(rbuf, rmax);
    k_attn  <<<NND / 16, 128, 0, stream>>>(A, sbuf, rbuf, rmax, HhT, out);
}
